// BiLevelRoutingAttention_32564442038680
// MI455X (gfx1250) — matrix operand order repaired and checked by reading, not compiled, not run on hardware
//
#include <hip/hip_runtime.h>
#include <hip/hip_bf16.h>

// ---------------------------------------------------------------------------
// Spiking bi-level routing attention for MI455X (gfx1250, wave32, WMMA).
// Pipeline (all f16 WMMA 16x16x32, f32 accum):
//   k_pack_w   : w_qkv/w_proj f32 -> f16, transposed (column-major) for B-frags
//   k_pack_xw  : window-partition gather x f32 -> xw f16 [N=65536, 256]
//   k_gemm_qkv : qkv f16 [N,768] = xw @ wqkv + b          (WMMA, 32x64 tiles)
//   k_region   : q/k region means [B,W,256]
//   k_route    : a_r = qr @ krT * scale, top-8 indices (lax.top_k tie rule)
//   k_lif      : LIF recurrence over T -> spikes; q natural layout,
//                k,v transposed per head [T,B,W,H,D,S] for contiguous frags
//   k_kv       : kv[h,d,e] = sum over routed tokens kT x vT   (WMMA, exact)
//   k_attn     : out = q @ kvT * scale                        (WMMA, exact in)
//   k_proj     : final = attn @ wproj + b, window-merge store (WMMA)
// ---------------------------------------------------------------------------

typedef _Float16 half_t;
typedef __attribute__((ext_vector_type(16))) _Float16 v16h;
typedef __attribute__((ext_vector_type(8)))  _Float16 v8h;
typedef __attribute__((ext_vector_type(8)))  float    v8f;

#define TDIM 4
#define BDIM 2
#define WDIM 32
#define SDIM 256
#define CDIM 256
#define HDS  8
#define DDIM 32
#define TOPK 8
#define NROWS (TDIM*BDIM*WDIM*SDIM)   /* 65536 */
#define C3    (3*CDIM)                /* 768   */
#define E22   (1 << 22)               /* B*W*S*C = 4194304 */
#define SCALE 0.17677669529663687f    /* 32^-0.5 */

// ---- WMMA fragment helpers (ISA 7.12.2 layouts) ---------------------------
// A-matrix 16x32 f16: lane m=l&15; lanes<16 hold K 0-7 & 16-23, lanes>=16
// hold K 8-15 & 24-31 (two contiguous 8-half groups).
__device__ __forceinline__ v16h fragA(const half_t* __restrict__ base,
                                      int stride, int lane) {
  int m  = lane & 15;
  int kh = (lane >> 4) * 8;
  const half_t* p = base + (size_t)m * stride + kh;
  v8h lo = *(const v8h*)(p);
  v8h hi = *(const v8h*)(p + 16);
  v16h r;
#pragma unroll
  for (int i = 0; i < 8; ++i) { r[i] = lo[i]; r[i + 8] = hi[i]; }
  return r;
}
// B-matrix 32x16 f16, stored column-major (stride = K-extent of the buffer):
// lane n=l&15; lanes<16 hold K 0-7 & 16-23, lanes>=16 hold K 8-15 & 24-31.
__device__ __forceinline__ v16h fragB(const half_t* __restrict__ base,
                                      int stride, int lane) {
  int n  = lane & 15;
  int kb = (lane >> 4) * 8;
  const half_t* p = base + (size_t)n * stride + kb;
  v8h lo = *(const v8h*)(p);
  v8h hi = *(const v8h*)(p + 16);
  v16h r;
#pragma unroll
  for (int i = 0; i < 8; ++i) { r[i] = lo[i]; r[i + 8] = hi[i]; }
  return r;
}
__device__ __forceinline__ v8f wmma_f16(v16h a, v16h b, v8f c) {
  return __builtin_amdgcn_wmma_f32_16x16x32_f16(false, a, false, b,
                                                (short)0, c, false, false);
}

// ---- window partition gather: x f32 -> xw16 [N,256] -----------------------
__global__ void k_pack_xw(const float* __restrict__ x,
                          half_t* __restrict__ xw) {
  int tid = blockIdx.x * blockDim.x + threadIdx.x;   // NROWS*CDIM/4 threads
  int r  = tid >> 6;
  int cq = (tid & 63) << 2;
  int s = r & 255, w = (r >> 8) & 31, b = (r >> 13) & 1, t = r >> 14;
  int iwt = w >> 4, iwh = (w >> 2) & 3, iww = w & 3;
  int igt = s >> 6, igh = (s >> 3) & 7, igw = s & 7;
  int Lt = iwt * 4 + igt, Lh = iwh * 8 + igh, Lw = iww * 8 + igw;
  size_t src = ((((size_t)(t * BDIM + b) * 8 + Lt) * 32 + Lh) * 32 + Lw) * CDIM + cq;
  float4 v = *(const float4*)(x + src);
  half_t* dst = xw + (size_t)r * CDIM + cq;
  dst[0] = (half_t)v.x; dst[1] = (half_t)v.y;
  dst[2] = (half_t)v.z; dst[3] = (half_t)v.w;
}

// ---- weights f32 -> f16 transposed ----------------------------------------
__global__ void k_pack_w(const float* __restrict__ wqkv,
                         const float* __restrict__ wproj,
                         half_t* __restrict__ wqkvT,
                         half_t* __restrict__ wprojT) {
  int tid = blockIdx.x * blockDim.x + threadIdx.x;   // (768+256)*256 threads
  if (tid < C3 * CDIM) {
    int n = tid >> 8, k = tid & 255;
    wqkvT[tid] = (half_t)wqkv[(size_t)k * C3 + n];
  } else {
    int u = tid - C3 * CDIM;
    int n = u >> 8, k = u & 255;
    wprojT[u] = (half_t)wproj[(size_t)k * CDIM + n];
  }
}

// ---- QKV GEMM: [N,768] = xw[N,256] @ wqkvT + b  (wave tile 32x64) ---------
__global__ void __launch_bounds__(256)
k_gemm_qkv(const half_t* __restrict__ A, const half_t* __restrict__ Bt,
           const float* __restrict__ bias, half_t* __restrict__ out) {
  int lane = threadIdx.x & 31, wave = threadIdx.x >> 5;
  int row0 = blockIdx.x * 256 + wave * 32;
  int col0 = blockIdx.y * 64;
  v8f acc[2][4] = {};
  for (int k0 = 0; k0 < CDIM; k0 += 32) {
    v16h a0 = fragA(A + (size_t)row0 * CDIM + k0,        CDIM, lane);
    v16h a1 = fragA(A + (size_t)(row0 + 16) * CDIM + k0, CDIM, lane);
#pragma unroll
    for (int tn = 0; tn < 4; ++tn) {
      v16h b = fragB(Bt + (size_t)(col0 + tn * 16) * CDIM + k0, CDIM, lane);
      acc[0][tn] = wmma_f16(a0, b, acc[0][tn]);
      acc[1][tn] = wmma_f16(a1, b, acc[1][tn]);
    }
  }
  int lrow = (lane >> 4) * 8, lcol = lane & 15;
#pragma unroll
  for (int tm = 0; tm < 2; ++tm)
#pragma unroll
    for (int tn = 0; tn < 4; ++tn) {
      int col = col0 + tn * 16 + lcol;
      float bv = bias[col];
#pragma unroll
      for (int j = 0; j < 8; ++j) {
        int row = row0 + tm * 16 + lrow + j;
        out[(size_t)row * C3 + col] = (half_t)(acc[tm][tn][j] + bv);
      }
    }
}

// ---- region means over (t,s) ----------------------------------------------
__global__ void k_region(const half_t* __restrict__ qkv,
                         float* __restrict__ qreg, float* __restrict__ kreg) {
  int bw = blockIdx.x;            // b*32 + w
  int c  = threadIdx.x;           // 256 threads
  int b = bw >> 5, w = bw & 31;
  float sq = 0.f, sk = 0.f;
  for (int t = 0; t < TDIM; ++t)
    for (int s = 0; s < SDIM; ++s) {
      size_t row = ((size_t)(t * BDIM + b) * WDIM + w) * SDIM + s;
      sq += (float)qkv[row * C3 + c];
      sk += (float)qkv[row * C3 + CDIM + c];
    }
  qreg[bw * CDIM + c] = sq * (1.0f / 1024.0f);
  kreg[bw * CDIM + c] = sk * (1.0f / 1024.0f);
}

// ---- routing scores + top-8 (ties -> lowest index, like lax.top_k) --------
__global__ void k_route(const float* __restrict__ qreg,
                        const float* __restrict__ kreg,
                        int* __restrict__ ridx) {
  __shared__ float sc[WDIM];
  int bw = blockIdx.x;            // 64 blocks
  int v  = threadIdx.x;           // 32 threads
  int b  = bw >> 5;
  float d = 0.f;
  for (int c = 0; c < CDIM; ++c)
    d += qreg[bw * CDIM + c] * kreg[(b * WDIM + v) * CDIM + c];
  sc[v] = d * SCALE;
  __syncthreads();
  if (v == 0) {
    bool used[WDIM];
    for (int u = 0; u < WDIM; ++u) used[u] = false;
    for (int j = 0; j < TOPK; ++j) {
      float best = -3.4e38f; int bi = 0;
      for (int u = 0; u < WDIM; ++u)
        if (!used[u] && sc[u] > best) { best = sc[u]; bi = u; }
      used[bi] = true;
      ridx[bw * TOPK + j] = bi;
    }
  }
}

// ---- LIF over T (tau=2, v_th=1, hard reset) -> spikes ---------------------
__global__ void k_lif(const half_t* __restrict__ qkv,
                      half_t* __restrict__ qs,   // [T][B*W*S][C]
                      half_t* __restrict__ kT,   // [T,B,W,H,D,S]
                      half_t* __restrict__ vT) { // [T,B,W,H,D,S]
  int tid = blockIdx.x * blockDim.x + threadIdx.x;   // 3 * 2^22 threads
  int which = tid >> 22;
  int e = tid & (E22 - 1);
  int c = e & 255, s = (e >> 8) & 255, w = (e >> 16) & 31, b = e >> 21;
  int h = c >> 5, d = c & 31;
  float vm = 0.f;
  for (int t = 0; t < TDIM; ++t) {
    size_t row = ((size_t)(t * BDIM + b) * WDIM + w) * SDIM + s;
    float xv = (float)qkv[row * C3 + which * CDIM + c];
    float hc = vm + (xv - vm) * 0.5f;
    float sp = (hc >= 1.0f) ? 1.0f : 0.0f;
    vm = hc * (1.0f - sp);
    half_t hs = (half_t)sp;
    if (which == 0) {
      qs[(size_t)t * E22 + e] = hs;
    } else {
      size_t idx = ((((size_t)(t * BDIM + b) * WDIM + w) * HDS + h) * DDIM + d) * SDIM + s;
      if (which == 1) kT[idx] = hs; else vT[idx] = hs;
    }
  }
}

// ---- kv = sum over routed windows of kT x vT : [32x32] per (t,b,w,h) ------
__global__ void __launch_bounds__(256)
k_kv(const half_t* __restrict__ kT, const half_t* __restrict__ vT,
     const int* __restrict__ ridx, half_t* __restrict__ kvT) {
  int lane = threadIdx.x & 31, wave = threadIdx.x >> 5;
  int unit = blockIdx.x * 8 + wave;               // 2048 = T*B*W*H
  int h = unit & 7, w = (unit >> 3) & 31, b = (unit >> 8) & 1, t = unit >> 9;
  const int* rid = ridx + (b * WDIM + w) * TOPK;
  v8f acc[4] = {};
  for (int j = 0; j < TOPK; ++j) {
    int wp = rid[j];
    size_t base = (((size_t)(t * BDIM + b) * WDIM + wp) * HDS + h) * DDIM * SDIM;
    const half_t* Ab = kT + base;                 // [D=32][S=256]
    const half_t* Bb = vT + base;                 // [E=32][S=256] (col-major)
#pragma unroll
    for (int kc = 0; kc < SDIM; kc += 32) {
      v16h a0 = fragA(Ab + kc,             SDIM, lane);
      v16h a1 = fragA(Ab + 16 * SDIM + kc, SDIM, lane);
      v16h b0 = fragB(Bb + kc,             SDIM, lane);
      v16h b1 = fragB(Bb + 16 * SDIM + kc, SDIM, lane);
      acc[0] = wmma_f16(a0, b0, acc[0]);
      acc[1] = wmma_f16(a0, b1, acc[1]);
      acc[2] = wmma_f16(a1, b0, acc[2]);
      acc[3] = wmma_f16(a1, b1, acc[3]);
    }
  }
  int lrow = (lane >> 4) * 8, lcol = lane & 15;
  half_t* outb = kvT + (size_t)unit * (DDIM * DDIM);  // stored [e][d]
#pragma unroll
  for (int tm = 0; tm < 2; ++tm)
#pragma unroll
    for (int tn = 0; tn < 2; ++tn)
#pragma unroll
      for (int j = 0; j < 8; ++j) {
        int d  = tm * 16 + lrow + j;
        int ec = tn * 16 + lcol;
        outb[ec * DDIM + d] = (half_t)acc[tm * 2 + tn][j];  // ints<=2048: exact
      }
}

// ---- out = q @ kvT * scale, f16 -> attn buffer (wave: 64 rows x 32 cols) --
__global__ void __launch_bounds__(256)
k_attn(const half_t* __restrict__ qs, const half_t* __restrict__ kvT,
       half_t* __restrict__ attn) {
  int lane = threadIdx.x & 31, wave = threadIdx.x >> 5;
  int unit = blockIdx.x * 8 + wave;               // 8192 = T*B*W*H*(S/64)
  int mt = unit & 3, h = (unit >> 2) & 7, w = (unit >> 5) & 31,
      b = (unit >> 10) & 1, t = unit >> 11;
  size_t rowbase = (size_t)((b * WDIM + w) * SDIM + mt * 64);
  const half_t* Ab = qs + (size_t)t * E22 + rowbase * CDIM + h * DDIM;
  const half_t* Bb = kvT + ((((size_t)(t * BDIM + b) * WDIM + w) * HDS + h)) * (DDIM * DDIM);
  v16h b0 = fragB(Bb,             DDIM, lane);    // kvT reused across 4 A frags
  v16h b1 = fragB(Bb + 16 * DDIM, DDIM, lane);
  v8f acc[4][2] = {};
#pragma unroll
  for (int i = 0; i < 4; ++i) {
    v16h a = fragA(Ab + (size_t)(i * 16) * CDIM, CDIM, lane);  // K = 32, 1 step
    acc[i][0] = wmma_f16(a, b0, acc[i][0]);
    acc[i][1] = wmma_f16(a, b1, acc[i][1]);
  }
  int lrow = (lane >> 4) * 8, lcol = lane & 15;
#pragma unroll
  for (int i = 0; i < 4; ++i)
#pragma unroll
    for (int j = 0; j < 8; ++j) {
      size_t row = (size_t)t * E22 / CDIM + rowbase + i * 16 + lrow + j;
      attn[row * CDIM + h * DDIM + lcol]      = (half_t)(acc[i][0][j] * SCALE);
      attn[row * CDIM + h * DDIM + 16 + lcol] = (half_t)(acc[i][1][j] * SCALE);
    }
}

// ---- proj GEMM + bias + window-merge scatter to d_out (f32) ---------------
__global__ void __launch_bounds__(256)
k_proj(const half_t* __restrict__ A, const half_t* __restrict__ Bt,
       const float* __restrict__ bias, float* __restrict__ out) {
  int lane = threadIdx.x & 31, wave = threadIdx.x >> 5;
  int row0 = blockIdx.x * 256 + wave * 32;
  int col0 = blockIdx.y * 64;
  v8f acc[2][4] = {};
  for (int k0 = 0; k0 < CDIM; k0 += 32) {
    v16h a0 = fragA(A + (size_t)row0 * CDIM + k0,        CDIM, lane);
    v16h a1 = fragA(A + (size_t)(row0 + 16) * CDIM + k0, CDIM, lane);
#pragma unroll
    for (int tn = 0; tn < 4; ++tn) {
      v16h b = fragB(Bt + (size_t)(col0 + tn * 16) * CDIM + k0, CDIM, lane);
      acc[0][tn] = wmma_f16(a0, b, acc[0][tn]);
      acc[1][tn] = wmma_f16(a1, b, acc[1][tn]);
    }
  }
  int lrow = (lane >> 4) * 8, lcol = lane & 15;
#pragma unroll
  for (int tm = 0; tm < 2; ++tm)
#pragma unroll
    for (int tn = 0; tn < 4; ++tn) {
      int col = col0 + tn * 16 + lcol;
      float bv = bias[col];
#pragma unroll
      for (int j = 0; j < 8; ++j) {
        int r = row0 + tm * 16 + lrow + j;
        int s = r & 255, w = (r >> 8) & 31, b = (r >> 13) & 1, t = r >> 14;
        int iwt = w >> 4, iwh = (w >> 2) & 3, iww = w & 3;
        int igt = s >> 6, igh = (s >> 3) & 7, igw = s & 7;
        int Lt = iwt * 4 + igt, Lh = iwh * 8 + igh, Lw = iww * 8 + igw;
        size_t dst = ((((size_t)(t * BDIM + b) * 8 + Lt) * 32 + Lh) * 32 + Lw) * CDIM + col;
        out[dst] = acc[tm][tn][j] + bv;
      }
    }
}

// ---------------------------------------------------------------------------
extern "C" void kernel_launch(void* const* d_in, const int* in_sizes, int n_in,
                              void* d_out, int out_size, void* d_ws, size_t ws_size,
                              hipStream_t stream) {
  const float* x      = (const float*)d_in[0];
  const float* w_qkv  = (const float*)d_in[1];
  const float* b_qkv  = (const float*)d_in[2];
  const float* w_proj = (const float*)d_in[3];
  const float* b_proj = (const float*)d_in[4];
  float* out = (float*)d_out;

  char* ws = (char*)d_ws;
  size_t off = 0;
  half_t* wqkvT  = (half_t*)(ws + off); off += (size_t)C3 * CDIM * 2;        // 384KB
  half_t* wprojT = (half_t*)(ws + off); off += (size_t)CDIM * CDIM * 2;      // 128KB
  half_t* xw16   = (half_t*)(ws + off); off += (size_t)NROWS * CDIM * 2;     // 32MB
  half_t* qkv16  = (half_t*)(ws + off); off += (size_t)NROWS * C3 * 2;       // 96MB
  half_t* qs     = (half_t*)(ws + off); off += (size_t)NROWS * CDIM * 2;     // 32MB
  half_t* kT     = (half_t*)(ws + off); off += (size_t)NROWS * CDIM * 2;     // 32MB
  half_t* vT     = (half_t*)(ws + off); off += (size_t)NROWS * CDIM * 2;     // 32MB
  half_t* kvT    = (half_t*)(ws + off); off += (size_t)TDIM*BDIM*WDIM*HDS*DDIM*DDIM*2; // 4MB
  half_t* attn   = (half_t*)(ws + off); off += (size_t)NROWS * CDIM * 2;     // 32MB
  float*  qreg   = (float*)(ws + off);  off += (size_t)BDIM*WDIM*CDIM*4;
  float*  kreg   = (float*)(ws + off);  off += (size_t)BDIM*WDIM*CDIM*4;
  int*    ridx   = (int*)(ws + off);    off += (size_t)BDIM*WDIM*TOPK*4;

  // 1. weights + input packing
  k_pack_w <<<((C3 + CDIM) * CDIM) / 256, 256, 0, stream>>>(w_qkv, w_proj, wqkvT, wprojT);
  k_pack_xw<<<(NROWS * CDIM / 4) / 256, 256, 0, stream>>>(x, xw16);
  // 2. QKV GEMM (WMMA)
  k_gemm_qkv<<<dim3(NROWS / 256, C3 / 64), 256, 0, stream>>>(xw16, wqkvT, b_qkv, qkv16);
  // 3. routing
  k_region<<<BDIM * WDIM, CDIM, 0, stream>>>(qkv16, qreg, kreg);
  k_route <<<BDIM * WDIM, WDIM, 0, stream>>>(qreg, kreg, ridx);
  // 4. LIF spikes
  k_lif<<<(3 * E22) / 256, 256, 0, stream>>>(qkv16, qs, kT, vT);
  // 5. linear attention (WMMA, exact on binary spikes)
  k_kv  <<<(TDIM * BDIM * WDIM * HDS) / 8, 256, 0, stream>>>(kT, vT, ridx, kvT);
  k_attn<<<(TDIM * BDIM * WDIM * HDS * (SDIM / 64)) / 8, 256, 0, stream>>>(qs, kvT, attn);
  // 6. projection GEMM + window merge (WMMA)
  k_proj<<<dim3(NROWS / 256, CDIM / 64), 256, 0, stream>>>(attn, wprojT, b_proj, out);
}
